// EGHG_13134009991420
// MI455X (gfx1250) — matrix operand order repaired and checked by reading, not compiled, not run on hardware
//
#include <hip/hip_runtime.h>

#define N_USERS 8000
#define M_ITEMS 8000
#define NTOT    16000
#define LAT     64
#define NLAYERS 3
#define BATCH   4096
#define KCHUNKS 500   // 16000 / 32

typedef __attribute__((ext_vector_type(16))) __bf16 v16bf;
typedef __attribute__((ext_vector_type(8)))  float  v8f;
typedef __attribute__((ext_vector_type(4)))  float  v4f;

// ---------------------------------------------------------------------------
// 1) all_emb = concat(user_emb, item_emb); acc = all_emb
// ---------------------------------------------------------------------------
__global__ void init_emb_kernel(const float* __restrict__ ue,
                                const float* __restrict__ ie,
                                float* __restrict__ all_emb,
                                float* __restrict__ acc) {
  int idx = blockIdx.x * blockDim.x + threadIdx.x;
  if (idx < NTOT * LAT) {
    float v = (idx < N_USERS * LAT) ? ue[idx] : ie[idx - N_USERS * LAT];
    all_emb[idx] = v;
    acc[idx] = v;
  }
}

// ---------------------------------------------------------------------------
// 2) h1 = all_emb @ W, emitted directly in the CDNA5 WMMA B-operand per-lane
//    layout as packed bf16 pairs.
//    Bp record for (kt, nt, lane): 8 dwords; dword j holds bf16 of
//    h1[kt*32 + 8*(lane>>4) + 16*(j>>2) + 2*(j&3) + {0,1}][nt*16 + (lane&15)]
// ---------------------------------------------------------------------------
__global__ void pack_kernel(const float* __restrict__ all_emb,
                            const float* __restrict__ W,
                            unsigned int* __restrict__ Bp) {
  int tid = blockIdx.x * blockDim.x + threadIdx.x;   // < 500*4*32*8 = 512000
  if (tid >= KCHUNKS * 4 * 32 * 8) return;
  int j    = tid & 7;
  int lane = (tid >> 3) & 31;
  int nt   = (tid >> 8) & 3;
  int kt   = tid >> 10;
  int col  = nt * 16 + (lane & 15);
  int k0   = kt * 32 + ((lane & 16) >> 1) + ((j & 4) << 2) + 2 * (j & 3);
  const float* r0 = all_emb + (size_t)k0 * LAT;
  const float* r1 = r0 + LAT;
  float d0 = 0.f, d1 = 0.f;
#pragma unroll
  for (int k = 0; k < LAT; ++k) {
    float w = W[k * LAT + col];
    d0 = fmaf(r0[k], w, d0);
    d1 = fmaf(r1[k], w, d1);
  }
  __bf16 b0 = (__bf16)d0, b1 = (__bf16)d1;
  unsigned int u0 = (unsigned int)__builtin_bit_cast(unsigned short, b0);
  unsigned int u1 = (unsigned int)__builtin_bit_cast(unsigned short, b1);
  Bp[((size_t)(kt * 4 + nt) * 32 + lane) * 8 + j] = u0 | (u1 << 16);
}

// ---------------------------------------------------------------------------
// 3) h2 = graph @ h1 ; all_emb = relu(h2 + bias) ; acc += all_emb
//    Block = 16 output rows x 64 cols. 8 waves split K=16000 (scalar bounds
//    via readfirstlane -> uniform scalar loop). A (HBM) is double-buffered in
//    registers: B loads (L2-hot) are issued first, then next-chunk A loads;
//    since VMEM loads complete in order, each chunk's A has a full iteration
//    of latency cover. Partials reduced through LDS, fused relu+bias.
// ---------------------------------------------------------------------------
__global__ __launch_bounds__(256) void gcn_gemm_kernel(
    const float* __restrict__ graph, const v16bf* __restrict__ Bp,
    const float* __restrict__ bias, float* __restrict__ all_emb,
    float* __restrict__ acc) {
  __shared__ float red[8 * 1024];

  const int wave  = __builtin_amdgcn_readfirstlane(threadIdx.x >> 5);
  const int lane  = threadIdx.x & 31;
  const int mrow  = blockIdx.x * 16 + (lane & 15);   // A row for this lane
  const int abase = (lane & 16) ? 8 : 0;             // A-layout K base offset

  // wave-uniform K-chunk range (500 chunks of 32 over 8 waves) — SGPR bounds
  const int c0 = (wave * KCHUNKS) >> 3;
  const int c1 = ((wave + 1) * KCHUNKS) >> 3;

  v8f cc[4] = {v8f{}, v8f{}, v8f{}, v8f{}};

  const float* arow = graph + (size_t)mrow * NTOT;

  // prologue: load A chunk c0
  int kb = c0 * 32 + abase;
  v4f a0 = __builtin_nontemporal_load((const v4f*)(arow + kb));
  v4f a1 = __builtin_nontemporal_load((const v4f*)(arow + kb + 4));
  v4f a2 = __builtin_nontemporal_load((const v4f*)(arow + kb + 16));
  v4f a3 = __builtin_nontemporal_load((const v4f*)(arow + kb + 20));

  for (int kt = c0; kt < c1; ++kt) {
    // (1) B tiles for this chunk (L2-resident packed layout)
    const v16bf* bp = Bp + ((size_t)kt << 7) + lane;  // [kt][nt][lane]
    v16bf b0 = bp[0];
    v16bf b1 = bp[32];
    v16bf b2 = bp[64];
    v16bf b3 = bp[96];

    // (2) next chunk's A from HBM (in-order completion => covered by next
    //     iteration's B wait); clamp so the tail re-reads the last chunk
    int kn = (kt + 1 < c1) ? (kt + 1) : kt;
    int kbn = kn * 32 + abase;
    v4f n0 = __builtin_nontemporal_load((const v4f*)(arow + kbn));
    v4f n1 = __builtin_nontemporal_load((const v4f*)(arow + kbn + 4));
    v4f n2 = __builtin_nontemporal_load((const v4f*)(arow + kbn + 16));
    v4f n3 = __builtin_nontemporal_load((const v4f*)(arow + kbn + 20));

    // (3) convert current A to the WMMA bf16 A layout
    v16bf a;
    a[0]  = (__bf16)a0[0]; a[1]  = (__bf16)a0[1];
    a[2]  = (__bf16)a0[2]; a[3]  = (__bf16)a0[3];
    a[4]  = (__bf16)a1[0]; a[5]  = (__bf16)a1[1];
    a[6]  = (__bf16)a1[2]; a[7]  = (__bf16)a1[3];
    a[8]  = (__bf16)a2[0]; a[9]  = (__bf16)a2[1];
    a[10] = (__bf16)a2[2]; a[11] = (__bf16)a2[3];
    a[12] = (__bf16)a3[0]; a[13] = (__bf16)a3[1];
    a[14] = (__bf16)a3[2]; a[15] = (__bf16)a3[3];

    // (4) 16x16x32 bf16 WMMAs over the full N=64
    cc[0] = __builtin_amdgcn_wmma_f32_16x16x32_bf16(false, a, false, b0,
                                                    (short)0, cc[0], false, false);
    cc[1] = __builtin_amdgcn_wmma_f32_16x16x32_bf16(false, a, false, b1,
                                                    (short)0, cc[1], false, false);
    cc[2] = __builtin_amdgcn_wmma_f32_16x16x32_bf16(false, a, false, b2,
                                                    (short)0, cc[2], false, false);
    cc[3] = __builtin_amdgcn_wmma_f32_16x16x32_bf16(false, a, false, b3,
                                                    (short)0, cc[3], false, false);

    // (5) rotate double buffer
    a0 = n0; a1 = n1; a2 = n2; a3 = n3;
  }

  // dump partial C tiles to LDS
#pragma unroll
  for (int t = 0; t < 4; ++t)
#pragma unroll
    for (int v = 0; v < 8; ++v)
      red[wave * 1024 + t * 256 + v * 32 + lane] = cc[t][v];
  __syncthreads();

  // reduce 8 wave-partials + fused relu(h + bias) epilogue
  for (int f = threadIdx.x; f < 1024; f += 256) {
    float s = 0.f;
#pragma unroll
    for (int w = 0; w < 8; ++w) s += red[w * 1024 + f];
    int t  = f >> 8;
    int v  = (f >> 5) & 7;
    int ln = f & 31;
    int row = blockIdx.x * 16 + v + ((ln & 16) ? 8 : 0);
    int col = t * 16 + (ln & 15);
    float e = s + bias[col];
    e = e > 0.f ? e : 0.f;
    size_t o = (size_t)row * LAT + col;
    all_emb[o] = e;
    acc[o] += e;
  }
}

// ---------------------------------------------------------------------------
// 4) gamma[b] = dot(acc[users[b]], acc[8000+items[b]]) / 16
// ---------------------------------------------------------------------------
__global__ void gamma_kernel(const int* __restrict__ users,
                             const int* __restrict__ items,
                             const float* __restrict__ acc,
                             float* __restrict__ out) {
  int b = blockIdx.x * blockDim.x + threadIdx.x;
  if (b >= BATCH) return;
  const v4f* pu = (const v4f*)(acc + (size_t)users[b] * LAT);
  const v4f* pi = (const v4f*)(acc + (size_t)(N_USERS + items[b]) * LAT);
  float s = 0.f;
#pragma unroll
  for (int k = 0; k < LAT / 4; ++k) {
    v4f u = pu[k], v = pi[k];
    s += u[0] * v[0] + u[1] * v[1] + u[2] * v[2] + u[3] * v[3];
  }
  out[b] = s * (1.0f / 16.0f);
}

// ---------------------------------------------------------------------------
extern "C" void kernel_launch(void* const* d_in, const int* in_sizes, int n_in,
                              void* d_out, int out_size, void* d_ws,
                              size_t ws_size, hipStream_t stream) {
  const int*   users = (const int*)d_in[0];
  const int*   items = (const int*)d_in[1];
  const float* graph = (const float*)d_in[2];
  const float* ue    = (const float*)d_in[3];
  const float* ie    = (const float*)d_in[4];
  const float* W     = (const float*)d_in[5];
  const float* bias  = (const float*)d_in[6];

  char* ws = (char*)d_ws;
  float*        acc     = (float*)(ws);                  // 4,096,000 B
  float*        all_emb = (float*)(ws + 4096000);        // 4,096,000 B
  unsigned int* Bp      = (unsigned int*)(ws + 8192000); // 2,048,000 B

  init_emb_kernel<<<(NTOT * LAT + 255) / 256, 256, 0, stream>>>(ue, ie, all_emb, acc);

  for (int layer = 0; layer < NLAYERS; ++layer) {
    pack_kernel<<<2000, 256, 0, stream>>>(all_emb, W, Bp);
    gcn_gemm_kernel<<<1000, 256, 0, stream>>>(graph, (const v16bf*)Bp, bias,
                                              all_emb, acc);
  }

  gamma_kernel<<<(BATCH + 255) / 256, 256, 0, stream>>>(users, items, acc,
                                                        (float*)d_out);
}
